// GatedGCN_37675453120557
// MI455X (gfx1250) — compile-verified, matrix operand order checked by reading, not run on hardware
//
#include <hip/hip_runtime.h>
#include <hip/hip_bf16.h>
#include <stdint.h>

// ---------------------------------------------------------------------------
// GatedGCN on MI455X (gfx1250), wave32.
//  - All dense GEMMs via v_wmma_f32_16x16x32_bf16 (bf16 in, f32 accum).
//  - Row-major A[M,K] and W[N,K]: the W^T B-fragment has the same per-lane
//    layout as the A-fragment (lane&15 selects row, K pairs striped), so each
//    fragment is two aligned b128 loads. K=128 -> 4 k-steps.
//  - Each wave computes a 16x64 strip (4 N-tiles): A fragment loaded once per
//    k-step, reused by 4 WMMAs -> 16 WMMAs/wave, 4x less A traffic.
//  - Edge messages kept bf16 (halves gather bytes); segment-sum via f32
//    global atomics (a/wh arrays are L2-resident: 192MB L2 >> 51MB arrays).
//  - GRU gates written bf16 straight from the GEMM epilogue.
// ---------------------------------------------------------------------------

typedef __bf16 bf16_t;
typedef __attribute__((ext_vector_type(16))) __bf16 v16bf;
typedef __attribute__((ext_vector_type(8)))  float  v8f;
typedef __attribute__((ext_vector_type(4)))  __bf16 v4bf;

#define NN      100000
#define NE      600000
#define FEATS   128
#define HID     128
#define ODIM    64
#define NSTEPS  3

// ---- f32 -> bf16 cast (vectorized x4) -------------------------------------
__global__ __launch_bounds__(256)
void cast_f32_bf16_kernel(const float* __restrict__ in, bf16_t* __restrict__ out, int n4)
{
    int i = blockIdx.x * 256 + threadIdx.x;
    if (i >= n4) return;
    float4 v = ((const float4*)in)[i];
    v4bf o;
    o[0] = (bf16_t)v.x; o[1] = (bf16_t)v.y; o[2] = (bf16_t)v.z; o[3] = (bf16_t)v.w;
    ((v4bf*)out)[i] = o;
}

// ---- WMMA GEMM: Y[M,N] = act(A[M,128] @ W[N,128]^T + bias) ----------------
// One wave per 16x64 strip (4 x 16x16 tiles), 8 waves per block.
// ACT: 0=none, 1=relu.  OUTF/OUTB select f32 / bf16 output streams.
template<int ACT, bool OUTF, bool OUTB>
__global__ __launch_bounds__(256)
void wmma_gemm_kernel(const bf16_t* __restrict__ A, const bf16_t* __restrict__ W,
                      const float* __restrict__ bias,
                      float* __restrict__ outF, bf16_t* __restrict__ outB,
                      int nGroups, int totalGroups, int N)
{
    const int grp = blockIdx.x * 8 + (threadIdx.x >> 5);    // wave-uniform
    if (grp >= totalGroups) return;                          // whole-wave exit
    const int mTile = grp / nGroups;
    const int nGrp  = grp - mTile * nGroups;                 // 64-col group

    const int lane = threadIdx.x & 31;
    const int half = lane >> 4;        // 0: K 0..7,16..23 ; 1: K 8..15,24..31
    const int lm   = lane & 15;        // row (A) / col (W)
    const int kb   = half << 3;

    const bf16_t* aP = A + (size_t)(mTile * 16 + lm) * 128 + kb;
    const bf16_t* wP = W + (size_t)(nGrp * 64 + lm) * 128 + kb;

    v8f acc[4] = {};
    #pragma unroll
    for (int kt = 0; kt < 128; kt += 32) {
        union { uint4 u[2]; v16bf v; } af;
        af.u[0] = *(const uint4*)(aP + kt);        // K = kt+kb .. +7
        af.u[1] = *(const uint4*)(aP + kt + 16);   // K = kt+16+kb .. +7
        #pragma unroll
        for (int j = 0; j < 4; ++j) {
            const bf16_t* wj = wP + (size_t)j * (16 * 128);
            union { uint4 u[2]; v16bf v; } bw;
            bw.u[0] = *(const uint4*)(wj + kt);
            bw.u[1] = *(const uint4*)(wj + kt + 16);
            acc[j] = __builtin_amdgcn_wmma_f32_16x16x32_bf16(
                         false, af.v, false, bw.v, (short)0, acc[j], false, false);
        }
    }

    const int row0 = mTile * 16 + (half << 3);     // C layout: VGPR d -> M=d(+8)
    #pragma unroll
    for (int j = 0; j < 4; ++j) {
        const int col = nGrp * 64 + j * 16 + lm;
        const float b = bias[col];
        #pragma unroll
        for (int d = 0; d < 8; ++d) {
            float v = acc[j][d] + b;
            if (ACT == 1) v = fmaxf(v, 0.0f);
            size_t off = (size_t)(row0 + d) * (size_t)N + col;
            if (OUTF) outF[off] = v;
            if (OUTB) outB[off] = (bf16_t)v;
        }
    }
}

// ---- edge gather + segment-sum scatter: a[dst] += wh[src] -----------------
// One wave per edge; lane covers 4 of 128 features (32*4 = 128).
__global__ __launch_bounds__(256)
void edge_scatter_kernel(const bf16_t* __restrict__ wh, const int* __restrict__ src,
                         const int* __restrict__ dst, float* __restrict__ a, int nEdges)
{
    int e = blockIdx.x * 8 + (threadIdx.x >> 5);
    if (e >= nEdges) return;
    int lane = threadIdx.x & 31;
    int s = src[e];
    int d = dst[e];
    v4bf m = ((const v4bf*)(wh + (size_t)s * HID))[lane];
    float* o = a + (size_t)d * HID + lane * 4;
    atomicAdd(o + 0, (float)m[0]);
    atomicAdd(o + 1, (float)m[1]);
    atomicAdd(o + 2, (float)m[2]);
    atomicAdd(o + 3, (float)m[3]);
}

// ---- fused GRU cell (biases already folded into gi/gh by GEMM epilogue) ---
__global__ __launch_bounds__(256)
void gru_kernel(const bf16_t* __restrict__ gi, const bf16_t* __restrict__ gh,
                float* __restrict__ h, bf16_t* __restrict__ hb, int total)
{
    int idx = blockIdx.x * 256 + threadIdx.x;
    if (idx >= total) return;
    int node = idx >> 7;        // /128
    int f    = idx & 127;
    size_t base = (size_t)node * 384;
    float ir  = (float)gi[base +        f];
    float iz  = (float)gi[base + 128 +  f];
    float inn = (float)gi[base + 256 +  f];
    float hr  = (float)gh[base +        f];
    float hz  = (float)gh[base + 128 +  f];
    float hn  = (float)gh[base + 256 +  f];
    float r = 1.0f / (1.0f + __expf(-(ir + hr)));
    float z = 1.0f / (1.0f + __expf(-(iz + hz)));
    float n = tanhf(inn + r * hn);
    float hv  = h[idx];
    float out = (1.0f - z) * n + z * hv;
    h[idx]  = out;
    hb[idx] = (bf16_t)out;
}

// ---------------------------------------------------------------------------
extern "C" void kernel_launch(void* const* d_in, const int* in_sizes, int n_in,
                              void* d_out, int out_size, void* d_ws, size_t ws_size,
                              hipStream_t stream)
{
    const float* x     = (const float*)d_in[0];
    const int*   src   = (const int*)  d_in[1];
    const int*   dst   = (const int*)  d_in[2];
    const float* W_in  = (const float*)d_in[3];
    const float* b_in  = (const float*)d_in[4];
    const float* W_e   = (const float*)d_in[5];
    const float* b_e   = (const float*)d_in[6];
    const float* W_ih  = (const float*)d_in[7];
    const float* b_ih  = (const float*)d_in[8];
    const float* W_hh  = (const float*)d_in[9];
    const float* b_hh  = (const float*)d_in[10];
    const float* W_out = (const float*)d_in[11];
    const float* b_out = (const float*)d_in[12];

    char* ws = (char*)d_ws;
    size_t off = 0;
    auto alloc = [&](size_t bytes) -> void* {
        void* p = ws + off;
        off = (off + bytes + 255) & ~(size_t)255;
        return p;
    };
    float*  h     = (float*) alloc((size_t)NN * HID * 4);
    bf16_t* h_bf  = (bf16_t*)alloc((size_t)NN * HID * 2);
    bf16_t* wh_bf = (bf16_t*)alloc((size_t)NN * HID * 2);
    float*  a     = (float*) alloc((size_t)NN * HID * 4);
    bf16_t* a_bf  = (bf16_t*)alloc((size_t)NN * HID * 2);  // also reused as x_bf
    bf16_t* gi_bf = (bf16_t*)alloc((size_t)NN * 3 * HID * 2);
    bf16_t* gh_bf = (bf16_t*)alloc((size_t)NN * 3 * HID * 2);
    bf16_t* Wb_in = (bf16_t*)alloc((size_t)HID * FEATS * 2);
    bf16_t* Wb_e  = (bf16_t*)alloc((size_t)HID * HID * 2);
    bf16_t* Wb_ih = (bf16_t*)alloc((size_t)3 * HID * HID * 2);
    bf16_t* Wb_hh = (bf16_t*)alloc((size_t)3 * HID * HID * 2);
    bf16_t* Wb_out= (bf16_t*)alloc((size_t)ODIM * HID * 2);

    auto cdiv = [](int x2, int y2) { return (x2 + y2 - 1) / y2; };
    auto cast = [&](const float* in, bf16_t* out, int n) {
        int n4 = n / 4;
        cast_f32_bf16_kernel<<<cdiv(n4, 256), 256, 0, stream>>>(in, out, n4);
    };

    // ---- weight / input casts (cheap; rerun every call for determinism) ----
    cast(W_in,  Wb_in,  HID * FEATS);
    cast(W_e,   Wb_e,   HID * HID);
    cast(W_ih,  Wb_ih,  3 * HID * HID);
    cast(W_hh,  Wb_hh,  3 * HID * HID);
    cast(W_out, Wb_out, ODIM * HID);
    cast(x,     a_bf,   NN * FEATS);      // x_bf aliases a_bf (a unused yet)

    const int mT = NN / 16;               // 6250, exact

    // ---- h = relu(x @ W_in^T + b_in), write f32 + bf16 ----
    {
        int nG = HID / 64, total = mT * nG;
        wmma_gemm_kernel<1, true, true><<<cdiv(total, 8), 256, 0, stream>>>(
            a_bf, Wb_in, b_in, h, h_bf, nG, total, HID);
    }

    for (int step = 0; step < NSTEPS; ++step) {
        // wh = h @ W_e^T + b_e  (bf16 only: consumed by gather)
        {
            int nG = HID / 64, total = mT * nG;
            wmma_gemm_kernel<0, false, true><<<cdiv(total, 8), 256, 0, stream>>>(
                h_bf, Wb_e, b_e, (float*)nullptr, wh_bf, nG, total, HID);
        }
        // a = segment_sum(wh[src], dst)
        hipMemsetAsync(a, 0, (size_t)NN * HID * 4, stream);
        edge_scatter_kernel<<<cdiv(NE, 8), 256, 0, stream>>>(wh_bf, src, dst, a, NE);
        cast(a, a_bf, NN * HID);
        // gi = a @ W_ih^T + b_ih ; gh = h @ W_hh^T + b_hh  (bf16 gates)
        {
            int nG = (3 * HID) / 64, total = mT * nG;
            wmma_gemm_kernel<0, false, true><<<cdiv(total, 8), 256, 0, stream>>>(
                a_bf, Wb_ih, b_ih, (float*)nullptr, gi_bf, nG, total, 3 * HID);
            wmma_gemm_kernel<0, false, true><<<cdiv(total, 8), 256, 0, stream>>>(
                h_bf, Wb_hh, b_hh, (float*)nullptr, gh_bf, nG, total, 3 * HID);
        }
        // h = GRU(a, h)
        gru_kernel<<<cdiv(NN * HID, 256), 256, 0, stream>>>(
            gi_bf, gh_bf, h, h_bf, NN * HID);
    }

    // ---- out = h @ W_out^T + b_out ----
    {
        int nG = ODIM / 64, total = mT * nG;
        wmma_gemm_kernel<0, true, false><<<cdiv(total, 8), 256, 0, stream>>>(
            h_bf, Wb_out, b_out, (float*)d_out, (bf16_t*)nullptr, nG, total, ODIM);
    }
}
